// SetEncoder_11175504904889
// MI455X (gfx1250) — matrix operand order repaired and checked by reading, not compiled, not run on hardware
//
#include <hip/hip_runtime.h>
#include <hip/hip_bf16.h>

typedef __bf16 bf16_t;
typedef __attribute__((ext_vector_type(16))) __bf16 v16bf;
typedef __attribute__((ext_vector_type(8)))  __bf16 v8bf;
typedef __attribute__((ext_vector_type(8)))  float   v8f;

#define NPTS    8192
#define IN_DIM  128
#define H_DIM   256
#define OUT_HALF 64
#define KNN     4

union FragU { v16bf v; v8bf h[2]; };

// ---------------------------------------------------------------------------
// WMMA fragment loaders (bf16, 16x16x32), per CDNA5 ISA 7.12.2 layouts.
// A (16x32, MxK), row-major source, ld = row stride (elements):
//   lanes 0-15 : M=lane,   K = {0..7} U {16..23}
//   lanes 16-31: M=lane-16,K = {8..15} U {24..31}
__device__ __forceinline__ v16bf load_a_frag(const bf16_t* __restrict__ base, int ld) {
    int lane = threadIdx.x & 31;
    const bf16_t* p = base + (lane & 15) * ld + ((lane < 16) ? 0 : 8);
    FragU f;
    f.h[0] = *(const v8bf*)p;          // first 8 K values
    f.h[1] = *(const v8bf*)(p + 16);   // +16 K offset chunk
    return f.v;
}
// B (32x16, KxN). Source arranged as Bt[n][k] (i.e. column n of B contiguous):
//   lanes 0-15 : N=lane,    K = {0..7} U {16..23}
//   lanes 16-31: N=lane-16, K = {8..15} U {24..31}
__device__ __forceinline__ v16bf load_b_frag(const bf16_t* __restrict__ base, int ld) {
    int lane = threadIdx.x & 31;
    const bf16_t* p = base + (lane & 15) * ld + ((lane < 16) ? 0 : 8);
    FragU f;
    f.h[0] = *(const v8bf*)p;
    f.h[1] = *(const v8bf*)(p + 16);
    return f.v;
}

__device__ __forceinline__ v8f wmma_bf16(v16bf a, v16bf b, v8f c) {
    return __builtin_amdgcn_wmma_f32_16x16x32_bf16(false, a, false, b, (short)0, c,
                                                   false, false);
}

// ---------------------------------------------------------------------------
// Elementwise f32 -> bf16 convert
__global__ __launch_bounds__(256)
void cvt_bf16_kernel(const float* __restrict__ in, bf16_t* __restrict__ out, int n) {
    int i = blockIdx.x * 256 + threadIdx.x;
    if (i < n) out[i] = (bf16_t)in[i];
}

// Transpose + convert: W[K][N] f32  ->  Wt[N][K] bf16
__global__ __launch_bounds__(256)
void transpose_bf16_kernel(const float* __restrict__ W, bf16_t* __restrict__ Wt,
                           int K, int N) {
    int i = blockIdx.x * 256 + threadIdx.x;
    if (i < K * N) {
        int k = i / N, n = i - k * N;
        Wt[n * K + k] = (bf16_t)W[i];
    }
}

// ---------------------------------------------------------------------------
// Tiled WMMA GEMM: C[M,N] = act(A[M,K] @ Bt[N,K]^T + bias)
// K / RELU / SPLIT are compile-time so the software-pipelined K loop fully
// unrolls: the cur<-nxt double-buffer rotation is SSA-renamed away (no
// v_mov chains) and loads for step k+1 stay in flight across step k's WMMAs.
// Block = 256 threads (8 waves). M-tile 64, N-tile 128; wave = 16x64 strip.
template <int K, int RELU, int SPLIT>
__global__ __launch_bounds__(256)
void gemm_bf16_kernel(const bf16_t* __restrict__ A, const bf16_t* __restrict__ Bt,
                      const float* __restrict__ bias,
                      float* __restrict__ Cf, bf16_t* __restrict__ Cb,
                      int M, int N) {
    constexpr int KSTEPS = K / 32;
    int wave = threadIdx.x >> 5;
    int lane = threadIdx.x & 31;
    int row0  = blockIdx.x * 64 + (wave & 3) * 16;
    int nbase = blockIdx.y * 128 + (wave >> 2) * 64;

    const bf16_t* Abase = A + (size_t)row0 * K;

    v8f acc[4] = {};
    v16bf a_cur = load_a_frag(Abase, K);
    v16bf b_cur[4];
#pragma unroll
    for (int t = 0; t < 4; ++t)
        b_cur[t] = load_b_frag(Bt + (size_t)(nbase + t * 16) * K, K);

#pragma unroll
    for (int ks = 0; ks < KSTEPS; ++ks) {
        v16bf a_nxt;
        v16bf b_nxt[4];
        if (ks + 1 < KSTEPS) {
            a_nxt = load_a_frag(Abase + (ks + 1) * 32, K);
#pragma unroll
            for (int t = 0; t < 4; ++t)
                b_nxt[t] = load_b_frag(Bt + (size_t)(nbase + t * 16) * K + (ks + 1) * 32, K);
        }
#pragma unroll
        for (int t = 0; t < 4; ++t)
            acc[t] = wmma_bf16(a_cur, b_cur[t], acc[t]);
        if (ks + 1 < KSTEPS) {
            a_cur = a_nxt;
#pragma unroll
            for (int t = 0; t < 4; ++t) b_cur[t] = b_nxt[t];
        }
    }

    int mrow = row0 + ((lane >> 4) << 3);     // C layout: VGPR r -> row mrow+r
#pragma unroll
    for (int t = 0; t < 4; ++t) {
        int n = nbase + t * 16 + (lane & 15);
        float bv = bias ? bias[n] : 0.0f;
#pragma unroll
        for (int r = 0; r < 8; ++r) {
            int m = mrow + r;
            float v = acc[t][r] + bv;
            if (RELU) v = fmaxf(v, 0.0f);
            if (SPLIT) {
                // final decoder layer: scatter into (out[:, :64], out[:, 64:])
                if (n < OUT_HALF) Cf[(size_t)m * OUT_HALF + n] = v;
                else Cf[(size_t)M * OUT_HALF + (size_t)m * OUT_HALF + (n - OUT_HALF)] = v;
            } else if (Cf) {
                Cf[(size_t)m * N + n] = v;
            }
            if (Cb) Cb[(size_t)m * N + n] = (bf16_t)v;
        }
    }
}

// ---------------------------------------------------------------------------
// Row squared norms: one wave per row of h[NPTS][H_DIM]
__global__ __launch_bounds__(256)
void rowsq_kernel(const float* __restrict__ h, float* __restrict__ sq) {
    int wave = threadIdx.x >> 5, lane = threadIdx.x & 31;
    int row = blockIdx.x * 8 + wave;
    const float* p = h + (size_t)row * H_DIM;
    float s = 0.0f;
    for (int j = lane; j < H_DIM; j += 32) { float v = p[j]; s += v * v; }
#pragma unroll
    for (int off = 16; off; off >>= 1) s += __shfl_xor(s, off, 32);
    if (lane == 0) sq[row] = s;
}

// ---------------------------------------------------------------------------
// Fused Gram-matrix + online top-4 kernel. Never materializes d[NxN].
// Block: 256 threads, 32 rows. Sweeps 8192 cols in 128-wide tiles.
//  - A fragments (this block's 32 rows) are invariant over the column sweep:
//    cached once in 64 VGPRs, zero A traffic in the hot loop.
//  - B fragments double-buffered across (fully unrolled) k-steps.
//  - next column tile prefetched via global_prefetch.
// Per tile: d = sq_r + sq_c - 2*S into LDS (stride 130, conflict-free)
// -> segmented per-thread top-4 -> final (value,index) stable merge.
__global__ __launch_bounds__(256)
void knn_topk_kernel(const bf16_t* __restrict__ hb, const float* __restrict__ sq,
                     int* __restrict__ idx_out, int* __restrict__ idx_dout) {
    __shared__ float smem[32 * 130];
    int tid  = threadIdx.x;
    int wave = tid >> 5, lane = tid & 31;
    int row0 = blockIdx.x * 32;
    int m16  = (wave & 1) * 16;      // this wave's row sub-block
    int nq   = (wave >> 1) * 32;     // this wave's 32-col strip

    int srow = tid & 31;             // scan assignment: row
    int scol = (tid >> 5) * 16;      // scan assignment: 16-col segment

    // cache the whole A panel for this wave (8 k-steps x 8 VGPRs = 64 VGPRs)
    v16bf a_frag[H_DIM / 32];
#pragma unroll
    for (int ks = 0; ks < H_DIM / 32; ++ks)
        a_frag[ks] = load_a_frag(hb + (size_t)(row0 + m16) * H_DIM + ks * 32, H_DIM);

    // row norms for this thread's epilogue rows (invariant over sweep)
    int mloc = m16 + ((lane >> 4) << 3);
    float sqr[8];
#pragma unroll
    for (int r = 0; r < 8; ++r) sqr[r] = sq[row0 + mloc + r];

    float bv[KNN] = {1e30f, 1e30f, 1e30f, 1e30f};
    int   bi[KNN] = {0, 0, 0, 0};

    for (int ct = 0; ct < NPTS / 128; ++ct) {
        int col0 = ct * 128;

        // prefetch next tile's B panel (hint only; 1 prefetch per thread)
        int nc = col0 + 128;
        if (nc < NPTS) {
            const bf16_t* pf = hb + (size_t)(nc + (tid >> 1)) * H_DIM + (tid & 1) * 128;
            __builtin_prefetch(pf, 0, 1);
        }

        const bf16_t* bbase0 = hb + (size_t)(col0 + nq) * H_DIM;
        const bf16_t* bbase1 = hb + (size_t)(col0 + nq + 16) * H_DIM;

        v8f acc[2] = {};
        v16bf b0 = load_b_frag(bbase0, H_DIM);
        v16bf b1 = load_b_frag(bbase1, H_DIM);
#pragma unroll
        for (int ks = 0; ks < H_DIM / 32; ++ks) {
            v16bf nb0, nb1;
            if (ks + 1 < H_DIM / 32) {
                nb0 = load_b_frag(bbase0 + (ks + 1) * 32, H_DIM);
                nb1 = load_b_frag(bbase1 + (ks + 1) * 32, H_DIM);
            }
            acc[0] = wmma_bf16(a_frag[ks], b0, acc[0]);
            acc[1] = wmma_bf16(a_frag[ks], b1, acc[1]);
            if (ks + 1 < H_DIM / 32) { b0 = nb0; b1 = nb1; }
        }

#pragma unroll
        for (int t = 0; t < 2; ++t) {
            int nloc = nq + t * 16 + (lane & 15);
            float sc = sq[col0 + nloc];
#pragma unroll
            for (int r = 0; r < 8; ++r) {
                float d = sqr[r] + sc - 2.0f * acc[t][r];
                smem[(mloc + r) * 130 + nloc] = d;
            }
        }
        __syncthreads();
        // online top-4 over this thread's 16-col segment
#pragma unroll 4
        for (int j = 0; j < 16; ++j) {
            float v = smem[srow * 130 + scol + j];
            if (v < bv[3]) {
                int gi = col0 + scol + j;
                int p = 3;
                while (p > 0 && v < bv[p - 1]) { bv[p] = bv[p - 1]; bi[p] = bi[p - 1]; --p; }
                bv[p] = v; bi[p] = gi;
            }
        }
        __syncthreads();
    }

    // merge 8 segment-local top-4 lists per row (reuse LDS)
    float* cv = smem;
    int*   ci = (int*)(smem + 1024);
    int seg = tid >> 5;
#pragma unroll
    for (int j = 0; j < KNN; ++j) {
        cv[srow * 32 + seg * 4 + j] = bv[j];
        ci[srow * 32 + seg * 4 + j] = bi[j];
    }
    __syncthreads();
    if (tid < 32) {
        float* v = cv + tid * 32;
        int*  ii = ci + tid * 32;
        int gid = (row0 + tid) * KNN;
#pragma unroll
        for (int s = 0; s < KNN; ++s) {
            float best = 1e30f; int bidx = 0x7fffffff; int bpos = 0;
            for (int j = 0; j < 32; ++j) {
                float vv = v[j]; int vi = ii[j];
                if (vv < best || (vv == best && vi < bidx)) { best = vv; bidx = vi; bpos = j; }
            }
            v[bpos] = 1e30f; ii[bpos] = 0x7fffffff;
            idx_out[gid + s]  = bidx;
            idx_dout[gid + s] = bidx;   // third reference output (raw int bits)
        }
    }
}

// ---------------------------------------------------------------------------
// Neighbor gather + mean/max pooling -> zc (bf16) [NPTS, 2*H_DIM]
__global__ __launch_bounds__(256)
void pool_kernel(const float* __restrict__ h, const int* __restrict__ idx,
                 bf16_t* __restrict__ zc) {
    int row = blockIdx.x;
    int c   = threadIdx.x;           // 0..255 = feature channel
    int base = row * KNN;
    float mu = 0.0f, mx = -1e30f;
#pragma unroll
    for (int j = 0; j < KNN; ++j) {
        float v = h[(size_t)idx[base + j] * H_DIM + c];
        mu += v; mx = fmaxf(mx, v);
    }
    mu *= (1.0f / KNN);
    zc[(size_t)row * (2 * H_DIM) + c]          = (bf16_t)mu;
    zc[(size_t)row * (2 * H_DIM) + H_DIM + c]  = (bf16_t)mx;
}

// ---------------------------------------------------------------------------
extern "C" void kernel_launch(void* const* d_in, const int* in_sizes, int n_in,
                              void* d_out, int out_size, void* d_ws, size_t ws_size,
                              hipStream_t stream) {
    const float* x  = (const float*)d_in[0];
    const float* W1 = (const float*)d_in[1];
    const float* b1 = (const float*)d_in[2];
    const float* W2 = (const float*)d_in[3];
    const float* b2 = (const float*)d_in[4];
    const float* W3 = (const float*)d_in[5];
    const float* b3 = (const float*)d_in[6];
    const float* W4 = (const float*)d_in[7];
    const float* b4 = (const float*)d_in[8];

    // workspace layout (all 256B aligned)
    char* ws = (char*)d_ws;
    size_t off = 0;
    auto alloc = [&](size_t bytes) { void* p = ws + off; off += (bytes + 255) & ~size_t(255); return p; };
    bf16_t* x_bf = (bf16_t*)alloc((size_t)NPTS * IN_DIM * 2);
    bf16_t* W1t  = (bf16_t*)alloc((size_t)IN_DIM * H_DIM * 2);       // [256][128]
    bf16_t* W2t  = (bf16_t*)alloc((size_t)H_DIM * H_DIM * 2);        // [256][256]
    bf16_t* W3t  = (bf16_t*)alloc((size_t)2*H_DIM * 2*H_DIM * 2);    // [512][512]
    bf16_t* W4t  = (bf16_t*)alloc((size_t)2*OUT_HALF * 2*H_DIM * 2); // [128][512]
    bf16_t* h1   = (bf16_t*)alloc((size_t)NPTS * H_DIM * 2);
    float*  h_f  = (float*) alloc((size_t)NPTS * H_DIM * 4);
    bf16_t* h_b  = (bf16_t*)alloc((size_t)NPTS * H_DIM * 2);
    float*  sq   = (float*) alloc((size_t)NPTS * 4);
    int*    idx  = (int*)   alloc((size_t)NPTS * KNN * 4);
    bf16_t* zc   = (bf16_t*)alloc((size_t)NPTS * 2*H_DIM * 2);
    bf16_t* g1   = (bf16_t*)alloc((size_t)NPTS * 2*H_DIM * 2);

    // input conversion / weight pre-transpose (tiny one-off work)
    cvt_bf16_kernel<<<(NPTS*IN_DIM + 255)/256, 256, 0, stream>>>(x, x_bf, NPTS*IN_DIM);
    transpose_bf16_kernel<<<(IN_DIM*H_DIM + 255)/256, 256, 0, stream>>>(W1, W1t, IN_DIM, H_DIM);
    transpose_bf16_kernel<<<(H_DIM*H_DIM + 255)/256, 256, 0, stream>>>(W2, W2t, H_DIM, H_DIM);
    transpose_bf16_kernel<<<(4*H_DIM*H_DIM + 255)/256, 256, 0, stream>>>(W3, W3t, 2*H_DIM, 2*H_DIM);
    transpose_bf16_kernel<<<(2*H_DIM*2*OUT_HALF + 255)/256, 256, 0, stream>>>(W4, W4t, 2*H_DIM, 2*OUT_HALF);

    // encoder: h = ReLU(x@W1+b1) @ W2 + b2
    gemm_bf16_kernel<IN_DIM, 1, 0><<<dim3(NPTS/64, H_DIM/128), 256, 0, stream>>>(
        x_bf, W1t, b1, nullptr, h1, NPTS, H_DIM);
    gemm_bf16_kernel<H_DIM, 0, 0><<<dim3(NPTS/64, H_DIM/128), 256, 0, stream>>>(
        h1, W2t, b2, h_f, h_b, NPTS, H_DIM);

    // squared norms + fused Gram/top-4 (writes idx to ws and to d_out tail)
    rowsq_kernel<<<NPTS/8, 256, 0, stream>>>(h_f, sq);
    int* idx_dout = (int*)d_out + 2 * (size_t)NPTS * OUT_HALF;
    knn_topk_kernel<<<NPTS/32, 256, 0, stream>>>(h_b, sq, idx, idx_dout);

    // gather + mean/max pool -> zc
    pool_kernel<<<NPTS, 256, 0, stream>>>(h_f, idx, zc);

    // decoder: out = ReLU(zc@W3+b3) @ W4 + b4, split into two 64-wide halves
    gemm_bf16_kernel<2 * H_DIM, 1, 0><<<dim3(NPTS/64, (2*H_DIM)/128), 256, 0, stream>>>(
        zc, W3t, b3, nullptr, g1, NPTS, 2*H_DIM);
    gemm_bf16_kernel<2 * H_DIM, 0, 1><<<dim3(NPTS/64, 1), 256, 0, stream>>>(
        g1, W4t, b4, (float*)d_out, nullptr, NPTS, 2*OUT_HALF);
}
